// AudioSNN_20976620273874
// MI455X (gfx1250) — matrix operand order repaired and checked by reading, not compiled, not run on hardware
//
#include <hip/hip_runtime.h>

// MI455X (gfx1250) fused SNN: 512 single-wave workgroups, each owns 16 batch
// rows for all 200 time steps. Layer-1/2 GEMMs via v_wmma_f32_16x16x32_bf16
// with the leaky membrane update fused into the WMMA accumulator chain.
// b1 folded in as an augmented K column; layer-2's operand transpose done
// with half-wave __shfl_xor of packed bf16 spike pairs. Per-step x tiles
// (16 rows x 160B, stride 32KB) are streamed by the Tensor Data Mover into
// double-buffered LDS, synchronized with s_wait_tensorcnt.

typedef __attribute__((ext_vector_type(16))) __bf16   v16bf;
typedef __attribute__((ext_vector_type(8)))  float    v8f;
typedef __attribute__((ext_vector_type(8)))  int      v8i;
typedef __attribute__((ext_vector_type(4)))  int      v4i;
typedef __attribute__((ext_vector_type(4)))  unsigned v4u;

static constexpr int   Bsz  = 8192;
static constexpr int   Tst  = 200;
static constexpr int   Fin  = 40;
static constexpr int   Hid  = 128;
static constexpr int   Cls  = 5;
static constexpr float BETA = 0.9f;
static constexpr float TH   = 1.0f;

#if defined(__has_builtin)
#if __has_builtin(__builtin_amdgcn_tensor_load_to_lds) && \
    __has_builtin(__builtin_amdgcn_s_wait_tensorcnt)
#define HAVE_TDM 1
#endif
#endif

__device__ __forceinline__ v8f wmma_bf16(v16bf a, v16bf b, v8f c) {
  // (neg_a, A, neg_b, B, c_mod, C, reuse_a, reuse_b)
  return __builtin_amdgcn_wmma_f32_16x16x32_bf16(false, a, false, b,
                                                 (short)0, c, false, false);
}

#if HAVE_TDM
// Issue one TDM descriptor: 2-D tile, 16 rows x 40 f32, row stride 8000 f32.
__device__ __forceinline__ void tdm_load_tile(const float* gsrc, unsigned lds_off) {
  unsigned long long ga = (unsigned long long)(uintptr_t)gsrc;
  // D# group0: count=1 | lds_addr | global_addr[56:0] | type=2
  v4u g0 = {1u, lds_off, (unsigned)ga,
            ((unsigned)(ga >> 32) & 0x01FFFFFFu) | 0x80000000u};
  // D# group1: data_size=4B; tensor_dim0=40; tensor_dim1=16; tile_dim0=40;
  //            tile_dim1=16; tensor_dim0_stride=8000
  v8i g1 = {0x00020000,            // workgroup_mask=0, data_size=2 (4B)
            (int)(40u << 16),      // tensor_dim0[15:0] @ bits 63:48
            (int)(16u << 16),      // tensor_dim0 hi=0, tensor_dim1[15:0]
            (int)(40u << 16),      // tensor_dim1 hi=0, tile_dim0=40
            16,                    // tile_dim1=16, tile_dim2=0
            8000,                  // tensor_dim0_stride[31:0]
            0, 0};                 // stride hi, tensor_dim1_stride=0
  v4i gz = {0, 0, 0, 0};
#if __clang_major__ >= 23
  v8i g4 = {0, 0, 0, 0, 0, 0, 0, 0};
  __builtin_amdgcn_tensor_load_to_lds(g0, g1, gz, gz, g4, 0);
#else
  __builtin_amdgcn_tensor_load_to_lds(g0, g1, gz, gz, 0);
#endif
}
#endif

__global__ __launch_bounds__(32)
void snn_fused_wmma(const float* __restrict__ x,
                    const float* __restrict__ w1,
                    const float* __restrict__ b1,
                    const float* __restrict__ w2,
                    const float* __restrict__ b2,
                    float* __restrict__ out) {
  const int  lane = threadIdx.x & 31;
  const int  bl   = lane & 15;   // batch column within tile / A-matrix row
  const int  kh   = lane >> 4;   // K-half selector for WMMA layouts
  const bool lo   = (kh == 0);
  const int  run1 = 16 * kh;     // A-layout first 8-K run offset
  const int  run2 = 16 * kh + 8; // A-layout second 8-K run offset
  const int  b    = blockIdx.x * 16 + bl;
  const int  b0   = blockIdx.x * 16;

#if HAVE_TDM
  __shared__ float xbuf[2][16 * Fin];  // double-buffered TDM destination
  const unsigned loff0 = (unsigned)(uintptr_t)(&xbuf[0][0]);
  const unsigned loff1 = (unsigned)(uintptr_t)(&xbuf[1][0]);
#endif

  // ---------------- W1 (+b1 as augmented K=40 column) in A-layout ----------
  v16bf w1a[8][2];
#pragma unroll
  for (int tt = 0; tt < 8; ++tt) {
    const int h = 16 * tt + bl;
    const float* wr = w1 + h * Fin;
    float c0[16];
    *reinterpret_cast<float4*>(c0 + 0)  = *reinterpret_cast<const float4*>(wr + run1);
    *reinterpret_cast<float4*>(c0 + 4)  = *reinterpret_cast<const float4*>(wr + run1 + 4);
    *reinterpret_cast<float4*>(c0 + 8)  = *reinterpret_cast<const float4*>(wr + run2);
    *reinterpret_cast<float4*>(c0 + 12) = *reinterpret_cast<const float4*>(wr + run2 + 4);
    v16bf a0;
#pragma unroll
    for (int j = 0; j < 16; ++j) a0[j] = (__bf16)c0[j];
    w1a[tt][0] = a0;
    float c1[8];
    *reinterpret_cast<float4*>(c1 + 0) = *reinterpret_cast<const float4*>(wr + 32);
    *reinterpret_cast<float4*>(c1 + 4) = *reinterpret_cast<const float4*>(wr + 36);
    const float bh = b1[h];
    v16bf a1;
#pragma unroll
    for (int j = 0; j < 8; ++j) {
      a1[j]     = (__bf16)(lo ? c1[j] : 0.0f);
      a1[8 + j] = (__bf16)((lo && j == 0) ? bh : 0.0f);
    }
    w1a[tt][1] = a1;
  }

  // ---------------- W2 (rows = class, padded to 16) in A-layout ------------
  v16bf w2a[4];
#pragma unroll
  for (int cc = 0; cc < 4; ++cc) {
    const int crow = (bl < Cls) ? bl : 0;  // keep loads in-bounds
    const float* wr = w2 + crow * Hid + 32 * cc;
    float c0[16];
    *reinterpret_cast<float4*>(c0 + 0)  = *reinterpret_cast<const float4*>(wr + run1);
    *reinterpret_cast<float4*>(c0 + 4)  = *reinterpret_cast<const float4*>(wr + run1 + 4);
    *reinterpret_cast<float4*>(c0 + 8)  = *reinterpret_cast<const float4*>(wr + run2);
    *reinterpret_cast<float4*>(c0 + 12) = *reinterpret_cast<const float4*>(wr + run2 + 4);
    v16bf a;
#pragma unroll
    for (int j = 0; j < 16; ++j) a[j] = (__bf16)((bl < Cls) ? c0[j] : 0.0f);
    w2a[cc] = a;
  }

  float b2s[8];
#pragma unroll
  for (int r = 0; r < 8; ++r) b2s[r] = (r < Cls && lo) ? b2[r] : 0.0f;

  // ---------------- persistent membranes -----------------------------------
  v8f m1[8];
#pragma unroll
  for (int tt = 0; tt < 8; ++tt) m1[tt] = v8f{};
  v8f m2 = v8f{};

  const float* xtile = x + (size_t)b0 * (Tst * Fin);  // tile base (row b0)
  const float* xrow  = x + (size_t)b  * (Tst * Fin);  // this lane's row
  (void)xrow;

#if HAVE_TDM
  tdm_load_tile(xtile, loff0);  // prologue: stage step 0
#endif

#pragma unroll 1
  for (int t = 0; t < Tst; ++t) {
    const float* xp;
#if HAVE_TDM
    if (t + 1 < Tst) {
      tdm_load_tile(xtile + (t + 1) * Fin, (t & 1) ? loff0 : loff1);
      __builtin_amdgcn_s_wait_tensorcnt(1);  // step-t tile complete (in-order)
    } else {
      __builtin_amdgcn_s_wait_tensorcnt(0);
    }
    xp = &xbuf[t & 1][0] + bl * Fin;         // LDS reads (ds_load_b128)
#else
    xp = xrow + t * Fin;
    if (t + 1 < Tst) {
      __builtin_prefetch(xrow + (t + 1) * Fin + 16 * kh, 0, 1);
      __builtin_prefetch(xrow + (t + 1) * Fin + 32, 0, 1);
    }
#endif

    // -------- x_t^T B-operand: chunk0 K=0..31, chunk1 K=32..63 (aug @40) ---
    float xa[16];
    {
      const float4* q = reinterpret_cast<const float4*>(xp + 16 * kh);
      *reinterpret_cast<float4*>(xa + 0)  = q[0];
      *reinterpret_cast<float4*>(xa + 4)  = q[1];
      *reinterpret_cast<float4*>(xa + 8)  = q[2];
      *reinterpret_cast<float4*>(xa + 12) = q[3];
    }
    v16bf xb0;
#pragma unroll
    for (int j = 0; j < 16; ++j) xb0[j] = (__bf16)xa[j];

    float xc[8];
    {
      const float4* q = reinterpret_cast<const float4*>(xp + 32);
      *reinterpret_cast<float4*>(xc + 0) = q[0];
      *reinterpret_cast<float4*>(xc + 4) = q[1];
    }
    v16bf xb1;
#pragma unroll
    for (int j = 0; j < 8; ++j) xb1[j] = lo ? (__bf16)xc[j] : (__bf16)0.0f;
    xb1[8] = lo ? (__bf16)1.0f : (__bf16)0.0f;  // aug column -> +b1
#pragma unroll
    for (int j = 9; j < 16; ++j) xb1[j] = (__bf16)0.0f;

    // -------- layer 1: m1 = beta*m1 - reset + W1@x^T + b1 (fused in C) -----
#pragma unroll
    for (int tt = 0; tt < 8; ++tt) {
      v8f cin;
#pragma unroll
      for (int r = 0; r < 8; ++r) {
        float m = m1[tt][r];
        cin[r] = BETA * m - ((m > TH) ? TH : 0.0f);
      }
      v8f acc = wmma_bf16(w1a[tt][0], xb0, cin);
      acc     = wmma_bf16(w1a[tt][1], xb1, acc);
      m1[tt]  = acc;
    }

    // -------- spikes: pack pairs as bf16{0,1}, half-wave swap for B-layout -
    int pk[8][4], pksw[8][4];
#pragma unroll
    for (int tt = 0; tt < 8; ++tt) {
#pragma unroll
      for (int p = 0; p < 4; ++p) {
        unsigned u = ((m1[tt][2 * p]     > TH) ? 0x00003F80u : 0u)
                   | ((m1[tt][2 * p + 1] > TH) ? 0x3F800000u : 0u);
        pk[tt][p]   = (int)u;
        pksw[tt][p] = __shfl_xor((int)u, 16, 32);
      }
    }

    // -------- layer 2: m2 = beta*m2 - reset + b2 + W2@spk1^T ---------------
    v8f acc2;
#pragma unroll
    for (int r = 0; r < 8; ++r) {
      float m = m2[r];
      acc2[r] = BETA * m - ((m > TH) ? TH : 0.0f) + b2s[r];
    }
#pragma unroll
    for (int c = 0; c < 4; ++c) {
      v8i bi;
#pragma unroll
      for (int v = 0; v < 4; ++v) bi[v]     = lo ? pk[2 * c][v]   : pksw[2 * c + 1][v];
#pragma unroll
      for (int v = 0; v < 4; ++v) bi[4 + v] = lo ? pksw[2 * c][v] : pk[2 * c + 1][v];
      acc2 = wmma_bf16(w2a[c], __builtin_bit_cast(v16bf, bi), acc2);
    }
    m2 = acc2;

    // -------- emit spk2 for the 5 valid classes ----------------------------
    if (lo) {
      float* op = out + (size_t)t * (Bsz * Cls) + (size_t)b * Cls;
#pragma unroll
      for (int r = 0; r < Cls; ++r) op[r] = (m2[r] > TH) ? 1.0f : 0.0f;
    }
  }
}

extern "C" void kernel_launch(void* const* d_in, const int* in_sizes, int n_in,
                              void* d_out, int out_size, void* d_ws, size_t ws_size,
                              hipStream_t stream) {
  (void)in_sizes; (void)n_in; (void)out_size; (void)d_ws; (void)ws_size;
  const float* x  = (const float*)d_in[0];
  const float* w1 = (const float*)d_in[1];
  const float* b1 = (const float*)d_in[2];
  const float* w2 = (const float*)d_in[3];
  const float* b2 = (const float*)d_in[4];
  float* out = (float*)d_out;
  dim3 grid(Bsz / 16);  // 512 one-wave workgroups = full batch parallelism
  dim3 block(32);
  snn_fused_wmma<<<grid, block, 0, stream>>>(x, w1, b1, w2, b2, out);
}
